// Quory_73675868996405
// MI455X (gfx1250) — matrix operand order repaired and checked by reading, not compiled, not run on hardware
//
#include <hip/hip_runtime.h>
#include <hip/hip_bf16.h>
#include <cstddef>
#include <cstdint>

// Problem constants
#define NB    32      // batches
#define NM    4096    // sequence (m)
#define ND    512     // input dims
#define NH    8       // heads
#define NHD   64      // head dim
#define NL    64      // query length (w)
#define NE2   128     // kv cols per head (k: 0..63, v: 64..127)
#define MT    64      // m tile

typedef __bf16 v16bf __attribute__((ext_vector_type(16)));
typedef float  v8f   __attribute__((ext_vector_type(8)));
typedef unsigned int v4u  __attribute__((ext_vector_type(4)));
typedef int          v4i_ __attribute__((ext_vector_type(4)));
typedef int          v8i_ __attribute__((ext_vector_type(8)));

#if defined(__has_builtin)
#if __has_builtin(__builtin_amdgcn_tensor_load_to_lds)
#define HAVE_TDM_LD 1
#if __clang_major__ >= 23
#warning "CDNA5 probe: tensor_load_to_lds path ENABLED (6-arg, clang>=23)"
#else
#warning "CDNA5 probe: tensor_load_to_lds path ENABLED (5-arg, clang<=22)"
#endif
#else
#warning "CDNA5 probe: tensor_load_to_lds NOT available - manual copy fallback"
#endif
#if __has_builtin(__builtin_amdgcn_tensor_store_from_lds)
#define HAVE_TDM_ST 1
#warning "CDNA5 probe: tensor_store_from_lds path ENABLED"
#else
#warning "CDNA5 probe: tensor_store_from_lds NOT available - manual store fallback"
#endif
#endif

// ---------------------------------------------------------------------------
// Kernel A: fused  kv = x @ W1[h]^T + b1  ->  flash attention over m tiles
// One workgroup per (h, b). 256 threads = 8 waves.
// ---------------------------------------------------------------------------
__global__ void __launch_bounds__(256)
fused_kv_attn_kernel(const float* __restrict__ x,
                     const float* __restrict__ q,
                     const float* __restrict__ fc_w,
                     const float* __restrict__ fc_b,
                     __bf16* __restrict__ newv)
{
    extern __shared__ char smem[];
    __bf16* w_lds  = (__bf16*)smem;              // [128][512]  W1 head slice
    __bf16* x_lds  = w_lds  + 128 * 512;         // [64][512]   x tile (m,k)
    __bf16* q_lds  = x_lds  + 64 * 512;          // [64][64]    q (w,hd)
    __bf16* k_lds  = q_lds  + 64 * 64;           // [64][64]    k tile (m,hd)
    __bf16* vt_lds = k_lds  + 64 * 64;           // [64][64]    v tile transposed (d,m)
    __bf16* p_lds  = vt_lds + 64 * 64;           // [64][64]    softmax probs (w,m)
    float*  s_lds  = (float*)(p_lds + 64 * 64);  // [64][64]    raw scores
    float*  rowmax = s_lds + 64 * 64;            // [64]
    float*  rowsum = rowmax + 64;                // [64]
    float*  corr   = rowsum + 64;                // [64]

    const int h    = blockIdx.x;
    const int b    = blockIdx.y;
    const int tid  = threadIdx.x;
    const int wave = tid >> 5;
    const int lane = tid & 31;
    const int col  = lane & 15;                  // C/B fragment column (N)
    const int rhalf = (lane < 16) ? 0 : 8;       // C fragment row offset
    const int abase = (lane < 16) ? 0 : 16;      // A fragment K base
    const int bbase = (lane < 16) ? 0 : 16;      // B fragment K base

    // Per-wave tile assignment (A row-block is wave-uniform so A fragments
    // are loaded once and reused across the wave's output tiles).
    const int rt  = wave >> 1;          // KV GEMM: m row-block (0..3)
    const int ct0 = (wave & 1) * 4;     // KV GEMM: first e' col-block (+0..3)
    const int wr  = wave >> 1;          // attention: w row-block (0..3)
    const int nc0 = (wave & 1) * 2;     // attention: first col-block (+0..1)

    // --- stage W1 head slice (rows h*128 .. h*128+127) as bf16 ---
    {
        const float* wsrc = fc_w + (size_t)h * NE2 * ND;
        for (int i = tid; i < NE2 * ND / 4; i += 256) {
            float4 f = ((const float4*)wsrc)[i];
            __bf16* d = w_lds + i * 4;
            d[0] = (__bf16)f.x; d[1] = (__bf16)f.y;
            d[2] = (__bf16)f.z; d[3] = (__bf16)f.w;
        }
    }
    // --- stage q for this head: q[w, h, hd] ---
    for (int i = tid; i < NL * NHD; i += 256) {
        int w = i >> 6, hd = i & 63;
        q_lds[i] = (__bf16)q[(w * NH + h) * NHD + hd];
    }
    if (tid < 64) { rowmax[tid] = -3.0e38f; rowsum[tid] = 0.0f; }

    // O accumulators: 16 (w,d) tiles of 16x16 over 8 waves -> 2 per wave
    v8f oacc[2];
#pragma unroll
    for (int t = 0; t < 2; ++t)
#pragma unroll
        for (int e = 0; e < 8; ++e) oacc[t][e] = 0.0f;

    const float* xbase = x + ((size_t)b * NM) * ND;

    for (int m0 = 0; m0 < NM; m0 += MT) {
        // ---- (1) stage x tile (64 x 512) fp32 -> bf16 ----
        __syncthreads();
        const float* xsrc = xbase + (size_t)m0 * ND;
        if (m0 + MT < NM)
            __builtin_prefetch(xbase + (size_t)(m0 + MT) * ND, 0, 1);
        for (int i = tid; i < MT * ND / 4; i += 256) {
            float4 f = ((const float4*)xsrc)[i];
            __bf16* d = x_lds + i * 4;
            d[0] = (__bf16)f.x; d[1] = (__bf16)f.y;
            d[2] = (__bf16)f.z; d[3] = (__bf16)f.w;
        }
        __syncthreads();

        // ---- (2) KV projection: C(64x128) = Xtile(64x512) * W1slice^T ----
        {
            v8f c[4];
#pragma unroll
            for (int t = 0; t < 4; ++t)
#pragma unroll
                for (int e = 0; e < 8; ++e) c[t][e] = 0.0f;
            const int arow = rt * 16 + col;
            for (int kk = 0; kk < ND; kk += 32) {
                v16bf af;
#pragma unroll
                for (int e = 0; e < 8; ++e) {
                    af[e]     = x_lds[arow * ND + kk + abase + e];
                    af[e + 8] = x_lds[arow * ND + kk + abase + 8 + e];
                }
#pragma unroll
                for (int tt = 0; tt < 4; ++tt) {
                    int bcol = (ct0 + tt) * 16 + col;
                    v16bf bfm;
#pragma unroll
                    for (int e = 0; e < 16; ++e)
                        bfm[e] = w_lds[bcol * ND + kk + bbase + e];
                    c[tt] = __builtin_amdgcn_wmma_f32_16x16x32_bf16(
                                false, af, false, bfm, (short)0, c[tt], false, false);
                }
            }
#pragma unroll
            for (int tt = 0; tt < 4; ++tt) {
                int ct = ct0 + tt;
                float bias = fc_b[h * NE2 + ct * 16 + col];
                if (ct < 4) {           // k part -> k_lds[m][hd]
#pragma unroll
                    for (int e = 0; e < 8; ++e) {
                        int m = rt * 16 + e + rhalf;
                        k_lds[m * NHD + ct * 16 + col] = (__bf16)(c[tt][e] + bias);
                    }
                } else {                // v part -> vt_lds[d][m] (transposed)
                    int d = (ct - 4) * 16 + col;
#pragma unroll
                    for (int e = 0; e < 8; ++e) {
                        int m = rt * 16 + e + rhalf;
                        vt_lds[d * MT + m] = (__bf16)(c[tt][e] + bias);
                    }
                }
            }
        }
        __syncthreads();

        // ---- (3) scores: S(64x64) = q(64x64) * k^T ----
        {
            v8f s[2];
#pragma unroll
            for (int t = 0; t < 2; ++t)
#pragma unroll
                for (int e = 0; e < 8; ++e) s[t][e] = 0.0f;
            const int arow = wr * 16 + col;
            for (int kk = 0; kk < NHD; kk += 32) {
                v16bf af;
#pragma unroll
                for (int e = 0; e < 8; ++e) {
                    af[e]     = q_lds[arow * NHD + kk + abase + e];
                    af[e + 8] = q_lds[arow * NHD + kk + abase + 8 + e];
                }
#pragma unroll
                for (int tt = 0; tt < 2; ++tt) {
                    int bcol = (nc0 + tt) * 16 + col;
                    v16bf bfm;
#pragma unroll
                    for (int e = 0; e < 16; ++e)
                        bfm[e] = k_lds[bcol * NHD + kk + bbase + e];
                    s[tt] = __builtin_amdgcn_wmma_f32_16x16x32_bf16(
                                false, af, false, bfm, (short)0, s[tt], false, false);
                }
            }
#pragma unroll
            for (int tt = 0; tt < 2; ++tt)
#pragma unroll
                for (int e = 0; e < 8; ++e)
                    s_lds[(wr * 16 + e + rhalf) * MT + (nc0 + tt) * 16 + col] = s[tt][e];
        }
        __syncthreads();

        // ---- (4) online softmax: 4 threads per w row, shfl_xor reduce ----
        {
            const int row = tid >> 2;        // 0..63
            const int sub = tid & 3;         // 16 cols each
            float mx   = rowmax[row];
            float tmax = mx;
#pragma unroll
            for (int j = 0; j < 16; ++j)
                tmax = fmaxf(tmax, s_lds[row * MT + sub * 16 + j]);
            tmax = fmaxf(tmax, __shfl_xor(tmax, 1, 32));
            tmax = fmaxf(tmax, __shfl_xor(tmax, 2, 32));
            float sum = 0.0f;
#pragma unroll
            for (int j = 0; j < 16; ++j) {
                float p = __expf(s_lds[row * MT + sub * 16 + j] - tmax);
                p_lds[row * MT + sub * 16 + j] = (__bf16)p;
                sum += p;
            }
            sum += __shfl_xor(sum, 1, 32);
            sum += __shfl_xor(sum, 2, 32);
            if (sub == 0) {
                float cr = __expf(mx - tmax);
                rowsum[row] = rowsum[row] * cr + sum;
                rowmax[row] = tmax;
                corr[row]   = cr;
            }
        }
        __syncthreads();

        // ---- (5) O = O*corr + P * V ----
        {
            const int arow = wr * 16 + col;
#pragma unroll
            for (int tt = 0; tt < 2; ++tt)
#pragma unroll
                for (int e = 0; e < 8; ++e)
                    oacc[tt][e] *= corr[wr * 16 + e + rhalf];
            for (int kk = 0; kk < MT; kk += 32) {
                v16bf af;
#pragma unroll
                for (int e = 0; e < 8; ++e) {
                    af[e]     = p_lds[arow * MT + kk + abase + e];
                    af[e + 8] = p_lds[arow * MT + kk + abase + 8 + e];
                }
#pragma unroll
                for (int tt = 0; tt < 2; ++tt) {
                    int bcol = (nc0 + tt) * 16 + col;   // output d
                    v16bf bfm;
#pragma unroll
                    for (int e = 0; e < 16; ++e)
                        bfm[e] = vt_lds[bcol * MT + kk + bbase + e];
                    oacc[tt] = __builtin_amdgcn_wmma_f32_16x16x32_bf16(
                                   false, af, false, bfm, (short)0, oacc[tt], false, false);
                }
            }
        }
        __syncthreads();
    }

    // ---- finalize: new_v[b, w, h*64 + d] = O / rowsum ----
#pragma unroll
    for (int tt = 0; tt < 2; ++tt) {
        int dc = nc0 + tt;
#pragma unroll
        for (int e = 0; e < 8; ++e) {
            int w = wr * 16 + e + rhalf;
            float val = oacc[tt][e] / rowsum[w];
            newv[((size_t)b * NL + w) * (NH * NHD) + h * NHD + dc * 16 + col] =
                (__bf16)val;
        }
    }
}

// ---------------------------------------------------------------------------
// Kernel B: out = new_v @ W2^T + b2, then LayerNorm over last dim (512).
// One workgroup per 16 output rows (2048 rows total -> 128 WGs).
// new_v staging via TDM load; final tile written back via TDM store.
// ---------------------------------------------------------------------------
__global__ void __launch_bounds__(256)
out_proj_ln_kernel(const __bf16* __restrict__ newv,
                   const float* __restrict__ fcfc_w,
                   const float* __restrict__ fcfc_b,
                   float* __restrict__ out)
{
    extern __shared__ char smem[];
    __bf16* a_lds = (__bf16*)smem;                 // [16][512]   new_v rows
    __bf16* w_lds = a_lds + 16 * 512;              // [512][32]   W2 k-slab
    float*  o_lds = (float*)(w_lds + 512 * 32);    // [16][512]   pre-LN / final
    float*  red   = o_lds + 16 * 512;              // [512]       reduce scratch
    float*  stats = red + 512;                     // [32] mean/inv

    const int rb   = blockIdx.x;                   // row block (16 rows)
    const int tid  = threadIdx.x;
    const int wave = tid >> 5;
    const int lane = tid & 31;
    const int col  = lane & 15;
    const int rhalf = (lane < 16) ? 0 : 8;
    const int abase = (lane < 16) ? 0 : 16;
    const int bbase = (lane < 16) ? 0 : 16;

#ifdef HAVE_TDM_LD
    // ---- TDM: async DMA of new_v rows [rb*16 .. rb*16+15] x 512 bf16 ----
    if (wave == 0) {
        unsigned long long gaddr =
            (unsigned long long)(uintptr_t)(newv + (size_t)rb * 16 * 512);
        v4u g0;
        g0[0] = 1u;                                    // count=1 (valid, user)
        g0[1] = (unsigned)(uintptr_t)a_lds;            // lds_addr (bytes)
        g0[2] = (unsigned)(gaddr & 0xffffffffu);       // global_addr[31:0]
        g0[3] = (unsigned)((gaddr >> 32) & 0x1ffffffu) // global_addr[56:32]
                | (2u << 30);                          // type = 2 ("image")
        v8i_ g1;
        g1[0] = (int)(1u << 16);                       // data_size = 2 bytes
        g1[1] = (int)(512u << 16);                     // tensor_dim0[15:0] = 512
        g1[2] = (int)(2048u << 16);                    // tensor_dim1[15:0] = 2048
        g1[3] = (int)(512u << 16);                     // tile_dim0 = 512
        g1[4] = 16;                                    // tile_dim1 = 16
        g1[5] = 512;                                   // tensor_dim0_stride
        g1[6] = 0;
        g1[7] = 0;
        v4i_ gz = {0, 0, 0, 0};
#if __clang_major__ >= 23
        v8i_ gz8 = {0, 0, 0, 0, 0, 0, 0, 0};
        __builtin_amdgcn_tensor_load_to_lds(g0, g1, gz, gz, gz8, 0);
#else
        __builtin_amdgcn_tensor_load_to_lds(g0, g1, gz, gz, 0);
#endif
        __builtin_amdgcn_s_wait_tensorcnt(0);
    }
#else
    for (int i = tid; i < 16 * 512; i += 256)
        a_lds[i] = newv[(size_t)rb * 16 * 512 + i];
#endif

    v8f acc[4];
#pragma unroll
    for (int t = 0; t < 4; ++t)
#pragma unroll
        for (int e = 0; e < 8; ++e) acc[t][e] = 0.0f;

    __syncthreads();

    for (int kk = 0; kk < 512; kk += 32) {
        // stage W2[:, kk:kk+32] as bf16: w_lds[d][j]
        for (int i = tid; i < 512 * 32; i += 256) {
            int d = i >> 5, j = i & 31;
            w_lds[i] = (__bf16)fcfc_w[d * 512 + kk + j];
        }
        __syncthreads();

        v16bf af;
        int arow = col;                 // only one 16-row block
#pragma unroll
        for (int e = 0; e < 8; ++e) {
            af[e]     = a_lds[arow * 512 + kk + abase + e];
            af[e + 8] = a_lds[arow * 512 + kk + abase + 8 + e];
        }
#pragma unroll
        for (int tt = 0; tt < 4; ++tt) {
            int d0 = wave * 64 + tt * 16 + col;     // output col d
            v16bf bfm;
#pragma unroll
            for (int e = 0; e < 16; ++e)
                bfm[e] = w_lds[d0 * 32 + bbase + e];
            acc[tt] = __builtin_amdgcn_wmma_f32_16x16x32_bf16(
                          false, af, false, bfm, (short)0, acc[tt], false, false);
        }
        __syncthreads();
    }

    // bias + park in LDS for LayerNorm
#pragma unroll
    for (int tt = 0; tt < 4; ++tt) {
        int d = wave * 64 + tt * 16 + col;
        float bias = fcfc_b[d];
#pragma unroll
        for (int e = 0; e < 8; ++e)
            o_lds[(e + rhalf) * 512 + d] = acc[tt][e] + bias;
    }
    __syncthreads();

    // LayerNorm: 16 threads per row
    const int row = tid >> 4, sub = tid & 15;
    float s = 0.0f, sq = 0.0f;
    for (int j = sub; j < 512; j += 16) {
        float v = o_lds[row * 512 + j];
        s += v; sq += v * v;
    }
    red[tid]       = s;
    red[256 + tid] = sq;
    __syncthreads();
    if (sub == 0) {
        float ts = 0.0f, tq = 0.0f;
        for (int j = 0; j < 16; ++j) {
            ts += red[row * 16 + j];
            tq += red[256 + row * 16 + j];
        }
        float mean = ts * (1.0f / 512.0f);
        float var  = tq * (1.0f / 512.0f) - mean * mean;
        stats[row]      = mean;
        stats[16 + row] = rsqrtf(var + 1e-5f);
    }
    __syncthreads();
    float mean = stats[row], inv = stats[16 + row];

#ifdef HAVE_TDM_ST
    // normalize in place, then TDM-store the 16x512 f32 tile to global
    for (int j = sub; j < 512; j += 16)
        o_lds[row * 512 + j] = (o_lds[row * 512 + j] - mean) * inv;
    __syncthreads();
    if (wave == 0) {
        unsigned long long gaddr =
            (unsigned long long)(uintptr_t)(out + (size_t)rb * 16 * 512);
        v4u g0;
        g0[0] = 1u;                                    // count=1 (valid, user)
        g0[1] = (unsigned)(uintptr_t)o_lds;            // lds_addr (bytes)
        g0[2] = (unsigned)(gaddr & 0xffffffffu);
        g0[3] = (unsigned)((gaddr >> 32) & 0x1ffffffu) | (2u << 30);
        v8i_ g1;
        g1[0] = (int)(2u << 16);                       // data_size = 4 bytes
        g1[1] = (int)(512u << 16);                     // tensor_dim0 = 512
        g1[2] = (int)(2048u << 16);                    // tensor_dim1 = 2048
        g1[3] = (int)(512u << 16);                     // tile_dim0 = 512
        g1[4] = 16;                                    // tile_dim1 = 16
        g1[5] = 512;                                   // tensor_dim0_stride
        g1[6] = 0;
        g1[7] = 0;
        v4i_ gz = {0, 0, 0, 0};
#if __clang_major__ >= 23
        v8i_ gz8 = {0, 0, 0, 0, 0, 0, 0, 0};
        __builtin_amdgcn_tensor_store_from_lds(g0, g1, gz, gz, gz8, 0);
#else
        __builtin_amdgcn_tensor_store_from_lds(g0, g1, gz, gz, 0);
#endif
        __builtin_amdgcn_s_wait_tensorcnt(0);
    }
#else
    size_t base = ((size_t)rb * 16 + row) * 512;
    for (int j = sub; j < 512; j += 16)
        out[base + j] = (o_lds[row * 512 + j] - mean) * inv;
#endif
}

// ---------------------------------------------------------------------------
extern "C" void kernel_launch(void* const* d_in, const int* in_sizes, int n_in,
                              void* d_out, int out_size, void* d_ws, size_t ws_size,
                              hipStream_t stream)
{
    (void)in_sizes; (void)n_in; (void)out_size; (void)ws_size;
    const float* x      = (const float*)d_in[0];
    const float* q      = (const float*)d_in[1];
    const float* fc_w   = (const float*)d_in[2];
    const float* fc_b   = (const float*)d_in[3];
    const float* fcfc_w = (const float*)d_in[4];
    const float* fcfc_b = (const float*)d_in[5];
    float*  out  = (float*)d_out;
    __bf16* newv = (__bf16*)d_ws;   // [32][64][512] bf16 = 2 MB

    const size_t smemA =
        (size_t)(128 * 512 + 64 * 512 + 4 * 64 * 64) * 2 +   // bf16 buffers
        (size_t)(64 * 64 + 3 * 64) * 4;                      // f32 buffers
    const size_t smemB =
        (size_t)(16 * 512 + 512 * 32) * 2 +
        (size_t)(16 * 512 + 512 + 32) * 4;

    hipFuncSetAttribute((const void*)fused_kv_attn_kernel,
                        hipFuncAttributeMaxDynamicSharedMemorySize, (int)smemA);
    hipFuncSetAttribute((const void*)out_proj_ln_kernel,
                        hipFuncAttributeMaxDynamicSharedMemorySize, (int)smemB);

    dim3 gridA(NH, NB);   // 8 x 32 = 256 workgroups
    fused_kv_attn_kernel<<<gridA, 256, smemA, stream>>>(x, q, fc_w, fc_b, newv);

    dim3 gridB(NB * NL / 16);   // 128 workgroups
    out_proj_ln_kernel<<<gridB, 256, smemB, stream>>>(newv, fcfc_w, fcfc_b, out);
}
